// GCNIIEncoder_86801289052367
// MI455X (gfx1250) — matrix operand order repaired and checked by reading, not compiled, not run on hardware
//
#include <hip/hip_runtime.h>
#include <math.h>

#define N_NODES 100000
#define E_EDGES 1200000
#define HID     128
#define LAYERS  8
#define ALPHA_C 0.2f
#define LN_EPS  1e-5f
#define LDSW    136   // padded LDS row stride (bf16 elems): 272B, 16B-aligned, bank stride 4

typedef __attribute__((ext_vector_type(16))) __bf16 v16bf;
typedef __attribute__((ext_vector_type(8)))  __bf16 v8bf;
typedef __attribute__((ext_vector_type(8)))  float  v8f;

__device__ __forceinline__ void fatomic_add(float* p, float v) {
    // relaxed, device-scope fp32 atomic -> hardware global_atomic_add_f32
    __hip_atomic_fetch_add(p, v, __ATOMIC_RELAXED, __HIP_MEMORY_SCOPE_AGENT);
}

// ---------------- degree / dinv ----------------
__global__ void k_init_deg(float* __restrict__ deg) {
    int i = blockIdx.x * blockDim.x + threadIdx.x;
    if (i < N_NODES) deg[i] = 1.0f;  // self loop
}

__global__ void k_accum_deg(const long long* __restrict__ dst, float* __restrict__ deg) {
    int e = blockIdx.x * blockDim.x + threadIdx.x;
    if (e < E_EDGES) fatomic_add(&deg[(int)dst[e]], 1.0f);
}

__global__ void k_dinv(float* __restrict__ deg) {
    int i = blockIdx.x * blockDim.x + threadIdx.x;
    if (i < N_NODES) deg[i] = rsqrtf(deg[i]);   // deg >= 1 always
}

// =============== projection GEMM: h = relu(x @ W + b); x0 = h; agg = dinv^2 * h ========
// 64 rows/block, 8 waves x 16 cols, 4 M-tiles per wave with B-fragment reuse.
__global__ __launch_bounds__(256) void k_proj(const float* __restrict__ x,
                                              const float* __restrict__ w,
                                              const float* __restrict__ bias,
                                              float* __restrict__ h,
                                              float* __restrict__ x0,
                                              const float* __restrict__ dinv,
                                              float* __restrict__ agg) {
    __shared__ __bf16 sWt[HID * LDSW];  // W transposed: sWt[n*LDSW + k]
    __shared__ __bf16 sA[64 * LDSW];    // 64 rows of A (bf16), padded stride
    const int tid = threadIdx.x;
    const int m0  = blockIdx.x * 64;

    // coalesced W read, transposed LDS write
    for (int i = tid; i < HID * HID; i += 256) {
        int k = i >> 7, n = i & 127;
        sWt[n * LDSW + k] = (__bf16)w[i];
    }
    for (int i = tid; i < 64 * HID; i += 256) {
        int rl = i >> 7, c = i & 127;
        int row = m0 + rl;
        sA[rl * LDSW + c] = (__bf16)(row < N_NODES ? x[(size_t)row * HID + c] : 0.0f);
    }
    __syncthreads();

    const int wave = tid >> 5, lane = tid & 31;
    const int hi = lane >> 4, lcol = lane & 15, arow = lane & 15;
    const int n0 = wave * 16;

    v8f acc[4] = {v8f{}, v8f{}, v8f{}, v8f{}};
#pragma unroll
    for (int kb = 0; kb < HID; kb += 32) {
        const int bbase = (n0 + lcol) * LDSW + kb + hi * 8;
        v8bf b0 = *(const v8bf*)&sWt[bbase];
        v8bf b1 = *(const v8bf*)&sWt[bbase + 16];
        v16bf b;
#pragma unroll
        for (int j = 0; j < 8; ++j) { b[j] = b0[j]; b[8 + j] = b1[j]; }
#pragma unroll
        for (int m = 0; m < 4; ++m) {
            const int abase = (m * 16 + arow) * LDSW + kb + hi * 8;
            v8bf a0 = *(const v8bf*)&sA[abase];
            v8bf a1 = *(const v8bf*)&sA[abase + 16];
            v16bf a;
#pragma unroll
            for (int j = 0; j < 8; ++j) { a[j] = a0[j]; a[8 + j] = a1[j]; }
            acc[m] = __builtin_amdgcn_wmma_f32_16x16x32_bf16(false, a, false, b,
                                                             (short)0, acc[m], false, false);
        }
    }

    const int col = n0 + lcol;
    const float bcol = bias[col];
#pragma unroll
    for (int m = 0; m < 4; ++m) {
#pragma unroll
        for (int r = 0; r < 8; ++r) {
            int row = m0 + m * 16 + r + 8 * hi;
            if (row < N_NODES) {
                float v = acc[m][r] + bcol;
                v = v > 0.0f ? v : 0.0f;
                size_t g = (size_t)row * HID + col;
                h[g]  = v;
                x0[g] = v;
                float d = dinv[row];
                agg[g] = d * d * v;     // self-loop init for layer 0 SpMM
            }
        }
    }
}

// ---------------- edge SpMM: agg[dst] += dinv[src]*dinv[dst]*h[src], one wave/edge -----
__global__ __launch_bounds__(256) void k_spmm(const float* __restrict__ h,
                                              const float* __restrict__ dinv,
                                              const long long* __restrict__ src,
                                              const long long* __restrict__ dst,
                                              float* __restrict__ agg) {
    int e = (int)((blockIdx.x * blockDim.x + threadIdx.x) >> 5);
    if (e >= E_EDGES) return;
    int lane = threadIdx.x & 31;
    int si = (int)src[e];
    int di = (int)dst[e];
    float w = dinv[si] * dinv[di];
    const float4 hv = *(const float4*)&h[(size_t)si * HID + lane * 4];
    float* ap = &agg[(size_t)di * HID + lane * 4];
    fatomic_add(ap + 0, w * hv.x);
    fatomic_add(ap + 1, w * hv.y);
    fatomic_add(ap + 2, w * hv.z);
    fatomic_add(ap + 3, w * hv.w);
}

// =============== layer GEMM: t = (1-beta)*s + beta*(s @ W), s = (1-a)*agg + a*x0 =======
__global__ __launch_bounds__(256) void k_layer(const float* __restrict__ agg,
                                               const float* __restrict__ x0,
                                               const float* __restrict__ w,
                                               float beta,
                                               float* __restrict__ t) {
    __shared__ __bf16 sWt[HID * LDSW];
    __shared__ __bf16 sA[64 * LDSW];
    const int tid = threadIdx.x;
    const int m0  = blockIdx.x * 64;

    for (int i = tid; i < HID * HID; i += 256) {
        int k = i >> 7, n = i & 127;
        sWt[n * LDSW + k] = (__bf16)w[i];
    }
    for (int i = tid; i < 64 * HID; i += 256) {
        int rl = i >> 7, c = i & 127;
        int row = m0 + rl;
        float sv = 0.0f;
        if (row < N_NODES) {
            size_t g = (size_t)row * HID + c;
            sv = (1.0f - ALPHA_C) * agg[g] + ALPHA_C * x0[g];
        }
        sA[rl * LDSW + c] = (__bf16)sv;
    }
    __syncthreads();

    const int wave = tid >> 5, lane = tid & 31;
    const int hi = lane >> 4, lcol = lane & 15, arow = lane & 15;
    const int n0 = wave * 16;

    v8f acc[4] = {v8f{}, v8f{}, v8f{}, v8f{}};
#pragma unroll
    for (int kb = 0; kb < HID; kb += 32) {
        const int bbase = (n0 + lcol) * LDSW + kb + hi * 8;
        v8bf b0 = *(const v8bf*)&sWt[bbase];
        v8bf b1 = *(const v8bf*)&sWt[bbase + 16];
        v16bf b;
#pragma unroll
        for (int j = 0; j < 8; ++j) { b[j] = b0[j]; b[8 + j] = b1[j]; }
#pragma unroll
        for (int m = 0; m < 4; ++m) {
            const int abase = (m * 16 + arow) * LDSW + kb + hi * 8;
            v8bf a0 = *(const v8bf*)&sA[abase];
            v8bf a1 = *(const v8bf*)&sA[abase + 16];
            v16bf a;
#pragma unroll
            for (int j = 0; j < 8; ++j) { a[j] = a0[j]; a[8 + j] = a1[j]; }
            acc[m] = __builtin_amdgcn_wmma_f32_16x16x32_bf16(false, a, false, b,
                                                             (short)0, acc[m], false, false);
        }
    }

    const int col = n0 + lcol;
#pragma unroll
    for (int m = 0; m < 4; ++m) {
#pragma unroll
        for (int r = 0; r < 8; ++r) {
            int row = m0 + m * 16 + r + 8 * hi;
            if (row < N_NODES) {
                size_t g = (size_t)row * HID + col;
                // recompute fp32 s at the output location (each location owned by
                // exactly one lane: read happens before that lane's own write,
                // so t may alias agg)
                float sv = (1.0f - ALPHA_C) * agg[g] + ALPHA_C * x0[g];
                t[g] = (1.0f - beta) * sv + beta * acc[m][r];
            }
        }
    }
}

// ---------------- LayerNorm (+ReLU) + fused next-layer agg init, one wave per row ------
__global__ __launch_bounds__(256) void k_ln(const float* __restrict__ t,
                                            const float* __restrict__ scale,
                                            const float* __restrict__ bias,
                                            float* __restrict__ out,
                                            const float* __restrict__ dinv,
                                            float* __restrict__ agg,
                                            int do_relu, int do_agg) {
    int row = (int)((blockIdx.x * blockDim.x + threadIdx.x) >> 5);
    if (row >= N_NODES) return;
    int lane = threadIdx.x & 31;
    int c = lane * 4;
    float4 v = *(const float4*)&t[(size_t)row * HID + c];
    float s = v.x + v.y + v.z + v.w;
    float q = v.x * v.x + v.y * v.y + v.z * v.z + v.w * v.w;
#pragma unroll
    for (int off = 16; off > 0; off >>= 1) {
        s += __shfl_xor(s, off, 32);
        q += __shfl_xor(q, off, 32);
    }
    float mu  = s * (1.0f / HID);
    float var = q * (1.0f / HID) - mu * mu;
    float rs  = rsqrtf(var + LN_EPS);
    float4 o;
    o.x = (v.x - mu) * rs * scale[c + 0] + bias[c + 0];
    o.y = (v.y - mu) * rs * scale[c + 1] + bias[c + 1];
    o.z = (v.z - mu) * rs * scale[c + 2] + bias[c + 2];
    o.w = (v.w - mu) * rs * scale[c + 3] + bias[c + 3];
    if (do_relu) {
        o.x = o.x > 0.f ? o.x : 0.f;
        o.y = o.y > 0.f ? o.y : 0.f;
        o.z = o.z > 0.f ? o.z : 0.f;
        o.w = o.w > 0.f ? o.w : 0.f;
    }
    *(float4*)&out[(size_t)row * HID + c] = o;
    if (do_agg) {   // self-loop init for the next layer's SpMM (t may alias agg)
        float d = dinv[row];
        float dd = d * d;
        float4 a4 = {dd * o.x, dd * o.y, dd * o.z, dd * o.w};
        *(float4*)&agg[(size_t)row * HID + c] = a4;
    }
}

extern "C" void kernel_launch(void* const* d_in, const int* in_sizes, int n_in,
                              void* d_out, int out_size, void* d_ws, size_t ws_size,
                              hipStream_t stream) {
    (void)in_sizes; (void)n_in; (void)out_size; (void)ws_size;
    const float*     x      = (const float*)d_in[0];
    const long long* ei     = (const long long*)d_in[1];   // int64 edge_index [2, E]
    const float*     lin_w  = (const float*)d_in[2];
    const float*     lin_b  = (const float*)d_in[3];
    const float*     conv_w = (const float*)d_in[4];
    const float*     ln_s   = (const float*)d_in[5];
    const float*     ln_b   = (const float*)d_in[6];

    float* h = (float*)d_out;  // h lives in d_out; final LayerNorm writes the answer

    char* ws = (char*)d_ws;
    float* dinv = (float*)ws;
    size_t off  = ((size_t)N_NODES * sizeof(float) + 255) & ~(size_t)255;
    float* x0   = (float*)(ws + off);
    float* agg  = x0 + (size_t)N_NODES * HID;

    const long long* src = ei;
    const long long* dst = ei + E_EDGES;

    k_init_deg <<<(N_NODES + 255) / 256, 256, 0, stream>>>(dinv);
    k_accum_deg<<<(E_EDGES + 255) / 256, 256, 0, stream>>>(dst, dinv);
    k_dinv     <<<(N_NODES + 255) / 256, 256, 0, stream>>>(dinv);

    k_proj<<<(N_NODES + 63) / 64, 256, 0, stream>>>(x, lin_w, lin_b, h, x0, dinv, agg);

    for (int l = 0; l < LAYERS; ++l) {
        float beta = logf(1.0f / (float)(l + 1) + 1.0f);
        k_spmm <<<((long long)E_EDGES * 32 + 255) / 256, 256, 0, stream>>>(h, dinv, src, dst, agg);
        k_layer<<<(N_NODES + 63) / 64, 256, 0, stream>>>(agg, x0,
                                                         conv_w + (size_t)l * HID * HID, beta, agg);
        int last = (l == LAYERS - 1);
        k_ln   <<<(N_NODES + 7) / 8, 256, 0, stream>>>(agg, ln_s + l * HID, ln_b + l * HID, h,
                                                       dinv, agg, !last, !last);
    }
}
